// Seq2seqLightModel_4836133175656
// MI455X (gfx1250) — matrix operand order repaired and checked by reading, not compiled, not run on hardware
//
#include <hip/hip_runtime.h>
#include <hip/hip_bf16.h>

// ---------------- problem constants (match reference) ----------------
constexpr int BATCH = 256;   // B
constexpr int ENCH  = 512;   // ENC (encoder hidden)
constexpr int WIDTH = 256;   // W  (encoder width / attention positions)
constexpr int ATTNS = 256;   // ATTN
constexpr int DECH  = 256;   // DEC
constexpr int EMBD  = 128;   // EMB
constexpr int VOC   = 81;    // V
constexpr int TLEN  = 100;   // text_max_len
constexpr int KFULL = EMBD + ENCH + DECH;   // 896 : [emb | context | h]
constexpr int GN    = 4 * DECH;             // 1024 : gate columns

// ---------------- WMMA / TDM types (gfx1250, wave32) ----------------
typedef __attribute__((ext_vector_type(16))) __bf16       v16bf;
typedef __attribute__((ext_vector_type(8)))  float        v8f;
typedef __attribute__((ext_vector_type(4)))  unsigned int v4u;
typedef __attribute__((ext_vector_type(8)))  int          v8i;
typedef __attribute__((ext_vector_type(4)))  int          v4i;

__device__ __forceinline__ v8f wmma_bf16(v16bf a, v16bf b, v8f c) {
  // (neg_a, A, neg_b, B, c_mod, C, reuse_a, reuse_b)
  return __builtin_amdgcn_wmma_f32_16x16x32_bf16(false, a, false, b,
                                                 (short)0, c, false, false);
}

// A-matrix (16x32 bf16) per-lane K offsets: lanes 0-15 hold K 0..15,
// lanes 16-31 hold K 16..31 (same k order as B)
__device__ __forceinline__ int ka_off(int j, int half) {
  return j + half * 16;
}
// B-matrix (32x16 bf16): lanes 0-15 hold K 0..15, lanes 16-31 hold K 16..31
__device__ __forceinline__ int kb_off(int j, int half) {
  return j + half * 16;
}

__device__ __forceinline__ float sigmoidf(float v) {
  return 1.0f / (1.0f + __expf(-v));
}

// ---------------- workspace layout (bytes) ----------------
constexpr size_t OFF_WE    = 0;                                        // f32 [B][ATTN][W]
constexpr size_t SZ_WE     = (size_t)BATCH * ATTNS * WIDTH * 4;        // 64 MB
constexpr size_t OFF_WT    = OFF_WE + SZ_WE;                           // bf16 [K/32][GN][32]
constexpr size_t SZ_WT     = (size_t)KFULL * GN * 2;
constexpr size_t OFF_AF    = OFF_WT + SZ_WT;                           // bf16 [B][KFULL]
constexpr size_t SZ_AF     = (size_t)BATCH * KFULL * 2;
constexpr size_t OFF_GATES = OFF_AF + SZ_AF;                           // f32 [B][GN]
constexpr size_t SZ_GATES  = (size_t)BATCH * GN * 4;
constexpr size_t OFF_H     = OFF_GATES + SZ_GATES;                     // f32 [B][DEC]
constexpr size_t SZ_H      = (size_t)BATCH * DECH * 4;
constexpr size_t OFF_C     = OFF_H + SZ_H;                             // f32 [B][DEC]
constexpr size_t SZ_C      = SZ_H;
constexpr size_t OFF_X     = OFF_C + SZ_C;                             // i32 [B]

// ============================================================================
// k_convert: pack [w_ih | w_hh] -> bf16 in a K-blocked layout
// WT[(k/32)*GN*32 + n*32 + k%32] so a B fragment (16 K-contiguous bf16 per
// lane) is one 32-byte contiguous run -> 2x b128 loads per lane per K-step.
// ============================================================================
__global__ __launch_bounds__(256) void k_convert(const float* __restrict__ w_ih,
                                                 const float* __restrict__ w_hh,
                                                 __bf16* __restrict__ WT) {
  int idx = blockIdx.x * blockDim.x + threadIdx.x;
  if (idx >= KFULL * GN) return;
  int k = idx / GN;
  int n = idx - k * GN;
  float v = (k < EMBD + ENCH) ? w_ih[(size_t)n * (EMBD + ENCH) + k]
                              : w_hh[(size_t)n * DECH + (k - (EMBD + ENCH))];
  WT[((size_t)(k >> 5) * GN + n) * 32 + (k & 31)] = (__bf16)v;
}

__global__ __launch_bounds__(256) void k_init(float* __restrict__ h,
                                              float* __restrict__ c,
                                              int* __restrict__ x,
                                              const int* __restrict__ sos) {
  int i = blockIdx.x * blockDim.x + threadIdx.x;
  if (i < BATCH * DECH) { h[i] = 0.0f; c[i] = 0.0f; }
  if (i < BATCH) x[i] = sos[0];
}

// ============================================================================
// k_weighted_enc: we[b] = enc_conv_w (ATTNxENC) @ enc_out[b] (ENCxW) + bias
// One wave computes a 16x64 strip: one A fragment feeds 4 WMMAs per K-step
// (4x A reuse, 4 independent accumulator chains for the XDL pipeline).
// ============================================================================
__global__ __launch_bounds__(32) void k_weighted_enc(
    const float* __restrict__ enc_out, const float* __restrict__ wc,
    const float* __restrict__ bc, float* __restrict__ we) {
  const int lane = threadIdx.x & 31;
  const int half = lane >> 4;
  const int ln16 = lane & 15;
  const int tn4 = blockIdx.x;         // 64-wide W strip (0..3)
  const int tm  = blockIdx.y;         // ATTN tile (0..15)
  const int b   = blockIdx.z;         // batch
  const float* arow  = wc + (size_t)(tm * 16 + ln16) * ENCH;
  const float* bbase = enc_out + (size_t)b * ENCH * WIDTH + tn4 * 64 + ln16;

  v8f acc0 = {}, acc1 = {}, acc2 = {}, acc3 = {};
#pragma unroll 2
  for (int k0 = 0; k0 < ENCH; k0 += 32) {
    v16bf af;
#pragma unroll
    for (int j = 0; j < 16; ++j) af[j] = (__bf16)arow[k0 + ka_off(j, half)];
    v16bf bf0, bf1, bf2, bf3;
#pragma unroll
    for (int j = 0; j < 16; ++j) {
      const float* br = bbase + (size_t)(k0 + kb_off(j, half)) * WIDTH;
      bf0[j] = (__bf16)br[0];
      bf1[j] = (__bf16)br[16];
      bf2[j] = (__bf16)br[32];
      bf3[j] = (__bf16)br[48];
    }
    acc0 = wmma_bf16(af, bf0, acc0);
    acc1 = wmma_bf16(af, bf1, acc1);
    acc2 = wmma_bf16(af, bf2, acc2);
    acc3 = wmma_bf16(af, bf3, acc3);
  }
  float* crow = we + (size_t)b * ATTNS * WIDTH + (size_t)tm * 16 * WIDTH + tn4 * 64;
#pragma unroll
  for (int r = 0; r < 8; ++r) {
    int m = r + half * 8;                       // ISA C/D layout: M = r + 8*half
    float bias = bc[tm * 16 + m];
    float* cr = crow + (size_t)m * WIDTH + ln16;
    cr[0]  = acc0[r] + bias;
    cr[16] = acc1[r] + bias;
    cr[32] = acc2[r] + bias;
    cr[48] = acc3[r] + bias;
  }
}

// ============================================================================
// k_attn_ctx: per batch row b (one block, 256 threads):
//   dec_proj = h @ dec_lin_w^T + b ; scores_w = sum_a tanh(we+dp)*ew + eb ;
//   softmax ; context = enc_out @ alpha (wave-parallel, lane-coalesced, shfl
//   reduce) ; build bf16 A-panel [emb|ctx|h] for the gates GEMM.
// we (64MB) + enc_out (128MB) together are exactly L2-sized -> L2 streaming.
// ============================================================================
__global__ __launch_bounds__(256) void k_attn_ctx(
    const float* __restrict__ enc_out, const float* __restrict__ we,
    const float* __restrict__ dlw, const float* __restrict__ dlb,
    const float* __restrict__ ew, const float* __restrict__ ebp,
    const float* __restrict__ emb, const int* __restrict__ x,
    const float* __restrict__ h, __bf16* __restrict__ Afull,
    float* __restrict__ out_attn, int t) {
  __shared__ float h_s[DECH];
  __shared__ float ew_s[ATTNS];
  __shared__ float dp[ATTNS];
  __shared__ float alpha[WIDTH];
  __shared__ float red[256];
  const int tid  = threadIdx.x;
  const int lane = tid & 31;
  const int wv   = tid >> 5;
  const int b    = blockIdx.x;

  h_s[tid]  = h[(size_t)b * DECH + tid];
  ew_s[tid] = ew[tid];
  __syncthreads();

  // dec_proj (attn row per thread, float4 row dot from L2-resident dec_lin_w)
  {
    const float4* wrow = (const float4*)(dlw + (size_t)tid * DECH);
    float s = dlb[tid];
#pragma unroll 4
    for (int d = 0; d < DECH / 4; ++d) {
      float4 w4 = wrow[d];
      s += h_s[4 * d] * w4.x + h_s[4 * d + 1] * w4.y +
           h_s[4 * d + 2] * w4.z + h_s[4 * d + 3] * w4.w;
    }
    dp[tid] = s;
  }
  __syncthreads();

  // energy scan down the attention axis; thread = position w (coalesced)
  const float* web = we + (size_t)b * ATTNS * WIDTH + tid;
  __builtin_prefetch(web, 0, 3);   // -> global_prefetch_b8
  float sc = ebp[0];
#pragma unroll 4
  for (int a = 0; a < ATTNS; ++a)
    sc += tanhf(web[(size_t)a * WIDTH] + dp[a]) * ew_s[a];

  // softmax over the 256 positions
  red[tid] = sc; __syncthreads();
  for (int o = 128; o > 0; o >>= 1) {
    if (tid < o) red[tid] = fmaxf(red[tid], red[tid + o]);
    __syncthreads();
  }
  const float mx = red[0]; __syncthreads();
  const float ex = __expf(sc - mx);
  red[tid] = ex; __syncthreads();
  for (int o = 128; o > 0; o >>= 1) {
    if (tid < o) red[tid] += red[tid + o];
    __syncthreads();
  }
  const float al = ex / red[0];
  alpha[tid] = al;
  out_attn[((size_t)b * TLEN + t) * WIDTH + tid] = al;
  __syncthreads();

  // build next-GEMM A panel in bf16: [ emb(x) | context | h ]
  __bf16* arow = Afull + (size_t)b * KFULL;
  if (tid < EMBD) arow[tid] = (__bf16)emb[(size_t)x[b] * EMBD + tid];
  arow[EMBD + ENCH + tid] = (__bf16)h_s[tid];

  // context: wave wv owns rows e = wv, wv+8, ...; lanes split w (coalesced),
  // wave32 shuffle-xor tree reduction.
  for (int e = wv; e < ENCH; e += 8) {
    const float* er = enc_out + ((size_t)b * ENCH + e) * WIDTH;
    float p = 0.0f;
#pragma unroll
    for (int w = lane; w < WIDTH; w += 32) p += er[w] * alpha[w];
#pragma unroll
    for (int o = 16; o > 0; o >>= 1) p += __shfl_xor(p, o, 32);
    if (lane == 0) arow[EMBD + e] = (__bf16)p;
  }
}

// ============================================================================
// k_gates: gates = Afull(256x896,bf16) @ WT(K-blocked, bf16) + b_ih + b_hh
// The 16xKFULL A panel for this M-tile is staged into LDS by the Tensor Data
// Mover (2-D D#: tile 896x16 @ 2B, contiguous strides), synchronized with
// s_wait_tensorcnt, then shared by all 4 waves (one 16x16 N-tile each).
// B fragments are 32-byte contiguous per lane thanks to the blocked layout.
// ============================================================================
__global__ __launch_bounds__(128) void k_gates(const __bf16* __restrict__ A,
                                               const __bf16* __restrict__ Bm,
                                               const float* __restrict__ b_ih,
                                               const float* __restrict__ b_hh,
                                               float* __restrict__ gates) {
  __shared__ __bf16 As[16 * KFULL];          // 28 KB of the 320 KB/WGP LDS
  const int tid  = threadIdx.x;
  const int lane = tid & 31;
  const int wv   = tid >> 5;
  const int half = lane >> 4;
  const int ln16 = lane & 15;
  const int tn = blockIdx.x * 4 + wv;        // N tile 0..63
  const int tm = blockIdx.y;                 // M tile 0..15

  if (wv == 0) {
    // ---- Tensor DMA descriptor (CDNA5 ISA) ----
    const unsigned long long gaddr =
        (unsigned long long)(uintptr_t)(A + (size_t)tm * 16 * KFULL);
    const unsigned int laddr = (unsigned int)(uintptr_t)(&As[0]); // LDS offset
    v4u g0 = {};
    v8i g1 = {};
    v4i g2 = {};   // tensor is 2-D: groups 2/3 unused (zero)
    v4i g3 = {};
    g0[0] = 1u;                                        // count=1, user mode
    g0[1] = laddr;                                     // lds_addr
    g0[2] = (unsigned int)(gaddr & 0xFFFFFFFFu);       // global_addr[31:0]
    g0[3] = (unsigned int)(gaddr >> 32) | (2u << 30);  // global_addr[56:32], type=2
    g1[0] = (int)(1u << 16);                           // data_size=1 (2 bytes)
    g1[1] = (int)((unsigned)KFULL << 16);              // tensor_dim0 = 896
    g1[2] = (int)(16u << 16);                          // tensor_dim1 = 16 rows
    g1[3] = (int)((unsigned)KFULL << 16);              // tile_dim0 = 896
    g1[4] = 16;                                        // tile_dim1 = 16
    g1[5] = KFULL;                                     // tensor_dim0_stride = 896
    g1[6] = 0;                                         // stride hi / dim1_stride
    g1[7] = 0;
#if defined(__clang_major__) && (__clang_major__ >= 23)
    v8i g4 = {};
    __builtin_amdgcn_tensor_load_to_lds(g0, g1, g2, g3, g4, 0);
#else
    __builtin_amdgcn_tensor_load_to_lds(g0, g1, g2, g3, 0);
#endif
    __builtin_amdgcn_s_wait_tensorcnt(0);
  }
  __syncthreads();

  const int col = tn * 16 + ln16;
  const __bf16* arow = &As[(size_t)ln16 * KFULL];

  v8f acc = {};
#pragma unroll 2
  for (int k0 = 0; k0 < KFULL; k0 += 32) {
    v16bf af, bf;
    // B: one contiguous 32-byte run per lane in the K-blocked layout
    const __bf16* bptr = Bm + ((size_t)(k0 >> 5) * GN + col) * 32 + half * 16;
#pragma unroll
    for (int j = 0; j < 16; ++j) {
      af[j] = arow[k0 + ka_off(j, half)];    // ds_load (b128-vectorizable)
      bf[j] = bptr[j];                       // global b128 pair
    }
    acc = wmma_bf16(af, bf, acc);
  }
  const float bias = b_ih[col] + b_hh[col];
  float* crow = gates + (size_t)tm * 16 * GN + tn * 16;
#pragma unroll
  for (int r = 0; r < 8; ++r) {
    int m = r + half * 8;
    crow[(size_t)m * GN + ln16] = acc[r] + bias;
  }
}

// ============================================================================
// k_lstm_out: LSTM pointwise (torch gate order i,f,g,o), logits, log-softmax,
// argmax -> greedy feedback token. One block per batch row.
// ============================================================================
__global__ __launch_bounds__(128) void k_lstm_out(
    const float* __restrict__ gates, const float* __restrict__ out_w,
    const float* __restrict__ out_b, float* __restrict__ h,
    float* __restrict__ c, int* __restrict__ x,
    float* __restrict__ out_logp, float* __restrict__ out_pred, int t) {
  __shared__ float h_s[DECH];
  __shared__ float lg[96];
  __shared__ float red[128];
  __shared__ int   redi[128];
  const int tid = threadIdx.x;
  const int b = blockIdx.x;
  const float* g = gates + (size_t)b * GN;

  for (int d = tid; d < DECH; d += 128) {
    float ig = sigmoidf(g[d]);
    float fg = sigmoidf(g[DECH + d]);
    float gg = tanhf(g[2 * DECH + d]);
    float og = sigmoidf(g[3 * DECH + d]);
    float cn = fg * c[(size_t)b * DECH + d] + ig * gg;
    float hn = og * tanhf(cn);
    c[(size_t)b * DECH + d] = cn;
    h[(size_t)b * DECH + d] = hn;
    h_s[d] = hn;
  }
  __syncthreads();

  float lv = -INFINITY;
  if (tid < VOC) {
    const float4* wr = (const float4*)(out_w + (size_t)tid * DECH);
    float s = out_b[tid];
#pragma unroll 4
    for (int d = 0; d < DECH / 4; ++d) {
      float4 w4 = wr[d];
      s += h_s[4 * d] * w4.x + h_s[4 * d + 1] * w4.y +
           h_s[4 * d + 2] * w4.z + h_s[4 * d + 3] * w4.w;
    }
    lg[tid] = s;
    lv = s;
  }
  red[tid] = lv; redi[tid] = tid;
  __syncthreads();
  for (int o = 64; o > 0; o >>= 1) {
    if (tid < o && red[tid + o] > red[tid]) {
      red[tid] = red[tid + o]; redi[tid] = redi[tid + o];
    }
    __syncthreads();
  }
  const float mx = red[0];
  const int   am = redi[0];
  __syncthreads();
  red[tid] = (tid < VOC) ? __expf(lg[tid] - mx) : 0.0f;
  __syncthreads();
  for (int o = 64; o > 0; o >>= 1) {
    if (tid < o) red[tid] += red[tid + o];
    __syncthreads();
  }
  const float lse = mx + __logf(red[0]);
  if (tid < VOC) out_logp[((size_t)b * TLEN + t) * VOC + tid] = lg[tid] - lse;
  if (tid == 0) {
    x[b] = am;
    out_pred[(size_t)b * TLEN + t] = (float)am;
  }
}

// ============================================================================
extern "C" void kernel_launch(void* const* d_in, const int* in_sizes, int n_in,
                              void* d_out, int out_size, void* d_ws, size_t ws_size,
                              hipStream_t stream) {
  const float* enc_out = (const float*)d_in[0];
  const float* emb     = (const float*)d_in[1];
  const float* wc      = (const float*)d_in[2];
  const float* bc      = (const float*)d_in[3];
  const float* dlw     = (const float*)d_in[4];
  const float* dlb     = (const float*)d_in[5];
  const float* ew      = (const float*)d_in[6];
  const float* eb      = (const float*)d_in[7];
  const float* w_ih    = (const float*)d_in[8];
  const float* w_hh    = (const float*)d_in[9];
  const float* b_ih    = (const float*)d_in[10];
  const float* b_hh    = (const float*)d_in[11];
  const float* out_w   = (const float*)d_in[12];
  const float* out_b   = (const float*)d_in[13];
  const int*   sos     = (const int*)d_in[15];

  char* ws = (char*)d_ws;
  float*   we    = (float*)(ws + OFF_WE);
  __bf16*  WT    = (__bf16*)(ws + OFF_WT);
  __bf16*  Afull = (__bf16*)(ws + OFF_AF);
  float*   gates = (float*)(ws + OFF_GATES);
  float*   hbuf  = (float*)(ws + OFF_H);
  float*   cbuf  = (float*)(ws + OFF_C);
  int*     xbuf  = (int*)(ws + OFF_X);

  float* out_logp = (float*)d_out;                                   // [B,T,V]
  float* out_pred = out_logp + (size_t)BATCH * TLEN * VOC;           // [B,T]
  float* out_attn = out_pred + (size_t)BATCH * TLEN;                 // [B,T,W]

  k_convert<<<(KFULL * GN + 255) / 256, 256, 0, stream>>>(w_ih, w_hh, WT);
  k_init<<<(BATCH * DECH + 255) / 256, 256, 0, stream>>>(hbuf, cbuf, xbuf, sos);
  k_weighted_enc<<<dim3(WIDTH / 64, ATTNS / 16, BATCH), 32, 0, stream>>>(
      enc_out, wc, bc, we);

  for (int t = 0; t < TLEN; ++t) {
    k_attn_ctx<<<BATCH, 256, 0, stream>>>(enc_out, we, dlw, dlb, ew, eb, emb,
                                          xbuf, hbuf, Afull, out_attn, t);
    k_gates<<<dim3(16, 16), 128, 0, stream>>>(Afull, WT, b_ih, b_hh, gates);
    k_lstm_out<<<BATCH, 128, 0, stream>>>(gates, out_w, out_b, hbuf, cbuf,
                                          xbuf, out_logp, out_pred, t);
  }
}
